// model_1_68186900791734
// MI455X (gfx1250) — matrix operand order repaired and checked by reading, not compiled, not run on hardware
//
#include <hip/hip_runtime.h>
#include <hip/hip_bf16.h>

typedef __attribute__((ext_vector_type(16))) __bf16 v16bf;
typedef __attribute__((ext_vector_type(2)))  __bf16 v2bf;
typedef __attribute__((ext_vector_type(8)))  float  v8f;

#define MDIM 256
#define BSZ  512
#define KTOT (MDIM * MDIM)  // 65536
#define NELT (BSZ * MDIM)   // 131072

// ---------------------------------------------------------------------------
// Kernel 1: y2[b,i] = (sigmoid(|X[b,i]|) @ W1^T + b1)^2
// ---------------------------------------------------------------------------
__global__ void k_y2(const float* __restrict__ X, const float* __restrict__ W1,
                     const float* __restrict__ b1, float* __restrict__ y2) {
    __shared__ float s[MDIM];
    const int b = blockIdx.x, t = threadIdx.x;
    const float x = X[b * MDIM + t];
    s[t] = 1.0f / (1.0f + __expf(-fabsf(x)));
    __syncthreads();
    float acc = b1[t];
    const float* w = W1 + (size_t)t * MDIM;
#pragma unroll 8
    for (int k = 0; k < MDIM; ++k) acc = fmaf(s[k], w[k], acc);
    y2[b * MDIM + t] = acc * acc;
}

// ---------------------------------------------------------------------------
// Kernel 2: batch moments of U[b,j,i] = (y2[b,i])^j folded into g[j], h[j]
//   g[j] = gamma[j] * rsqrt(var[j] + eps), h[j] = beta[j] - mean[j]*g[j]
// Each block owns 8 consecutive j's and scans all 131072 elements.
// ---------------------------------------------------------------------------
__global__ void k_moments(const float* __restrict__ y2,
                          const float* __restrict__ gamma_, const float* __restrict__ beta_,
                          float* __restrict__ g, float* __restrict__ h) {
    const int j0 = blockIdx.x * 8;
    const int t = threadIdx.x;
    float s1[8], s2[8];
#pragma unroll
    for (int k = 0; k < 8; ++k) { s1[k] = 0.f; s2[k] = 0.f; }

    for (int e = t; e < NELT; e += 256) {
        const float v = y2[e];
        // p = v^j0 by binary exponentiation (j0 uniform per block)
        float p = 1.0f, base = v;
        int ex = j0;
        while (ex) { if (ex & 1) p *= base; base *= base; ex >>= 1; }
#pragma unroll
        for (int k = 0; k < 8; ++k) { s1[k] += p; s2[k] += p * p; p *= v; }
    }

    __shared__ float red[256];
    __shared__ float o1[8], o2[8];
    for (int k = 0; k < 8; ++k) {
        red[t] = s1[k]; __syncthreads();
        for (int off = 128; off; off >>= 1) { if (t < off) red[t] += red[t + off]; __syncthreads(); }
        if (t == 0) o1[k] = red[0];
        __syncthreads();
        red[t] = s2[k]; __syncthreads();
        for (int off = 128; off; off >>= 1) { if (t < off) red[t] += red[t + off]; __syncthreads(); }
        if (t == 0) o2[k] = red[0];
        __syncthreads();
    }
    if (t < 8) {
        const int j = j0 + t;
        const float inv_n = 1.0f / (float)NELT;
        const float mean = o1[t] * inv_n;
        const float var  = o2[t] * inv_n - mean * mean;
        const float gj   = gamma_[j] * rsqrtf(var + 1e-5f);
        g[j] = gj;
        h[j] = beta_[j] - mean * gj;
    }
}

// ---------------------------------------------------------------------------
// Kernel 3: W2t[k][o] = bf16(W2[o][k])  (k-major bf16 copy for WMMA B loads)
// 32k x 256o tile through padded LDS; coalesced reads and writes.
// ---------------------------------------------------------------------------
__global__ void k_w2t(const float* __restrict__ W2, __bf16* __restrict__ W2t) {
    __shared__ float tile[32 * 258];
    const int k0 = blockIdx.x * 32;
    const int t = threadIdx.x;
    const int kk = t & 31, og = t >> 5;                 // read: 32 lanes along k, 8 o's/iter
    for (int o = og; o < MDIM; o += 8)
        tile[kk * 258 + o] = W2[(size_t)o * KTOT + k0 + kk];
    __syncthreads();
    const int kw = t >> 3, oc = (t & 7) * 32;           // write: 8 threads per k-row
    for (int o = oc; o < oc + 32; ++o)
        W2t[(size_t)(k0 + kw) * MDIM + o] = (__bf16)tile[kw * 258 + o];
}

// ---------------------------------------------------------------------------
// Kernel 4: Kc[o] = sum_k h[k>>8] * W2[o][k] + b2[o]   (BN shift folded out)
// ---------------------------------------------------------------------------
__global__ void k_kc(const float* __restrict__ W2, const float* __restrict__ h,
                     const float* __restrict__ b2, float* __restrict__ Kc) {
    __shared__ float red[256];
    const int o = blockIdx.x, t = threadIdx.x;
    const float* row = W2 + (size_t)o * KTOT;
    float acc = 0.f;
    for (int k = t; k < KTOT; k += 256) acc = fmaf(h[k >> 8], row[k], acc);
    red[t] = acc; __syncthreads();
    for (int off = 128; off; off >>= 1) { if (t < off) red[t] += red[t + off]; __syncthreads(); }
    if (t == 0) Kc[o] = red[0] + b2[o];
}

// ---------------------------------------------------------------------------
// Kernel 5: the big WMMA GEMM.
//   c[b,o] = sum_{j,i} (g[j] * y2[b,i]^j) * W2t[j*256+i][o] + Kc[o]
// Workgroup: 32 b-rows x 128 o-cols, 8 waves, 2 accumulator tiles each.
// Running powers P[b,i] = y2^j live in registers; A fragments are staged in
// LDS in the exact CDNA5 16-bit A-matrix layout (lane=M, half-pairs=K).
// ---------------------------------------------------------------------------
__global__ void __launch_bounds__(256) k_gemm(
    const float* __restrict__ y2, const float* __restrict__ g,
    const __bf16* __restrict__ W2t, const float* __restrict__ Kc,
    float* __restrict__ cbuf) {
    // [m_t(2)][kc(8)][lane(32)][half(16)] = 8192 bf16 = 16 KB
    __shared__ __attribute__((aligned(32))) __bf16 Afrag[2 * 8 * 32 * 16];

    const int t    = threadIdx.x;
    const int lane = t & 31;
    const int wave = t >> 5;
    const int o0   = blockIdx.x * 128;
    const int b0   = blockIdx.y * 32;

    // A-generation ownership: thread owns row r = t&31, i-chunk kc_own = t>>5
    const int r      = t & 31;
    const int kc_own = t >> 5;
    const int m_t    = r >> 4;
    const int mloc   = r & 15;

    float y2r[32], P[32];
    const float* ybase = y2 + (size_t)(b0 + r) * MDIM + kc_own * 32;
#pragma unroll
    for (int e = 0; e < 32; ++e) { y2r[e] = ybase[e]; P[e] = 1.0f; }

    // wave -> output tiles: m-tile wm, n-tiles wn0, wn0+1
    const int wm  = wave & 1;
    const int wn0 = (wave >> 1) * 2;
    v8f acc0 = {}; v8f acc1 = {};

    for (int j = 0; j < MDIM; ++j) {
        const float gj = g[j];
        if (j > 0) {
#pragma unroll
            for (int e = 0; e < 32; ++e) P[e] *= y2r[e];   // y2^j
        }
        // Stage A fragment values in ISA A-layout; paired ds_store_b32 writes.
#pragma unroll
        for (int e = 0; e < 32; e += 2) {
            const int lane_w = mloc + ((e & 8) ? 16 : 0);
            const int hh     = (e & 16) ? (8 + (e & 7)) : (e & 7);
            const int slot   = (((m_t * 8 + kc_own) * 32) + lane_w) * 16 + hh;
            v2bf pr = { (__bf16)(gj * P[e]), (__bf16)(gj * P[e + 1]) };
            *reinterpret_cast<v2bf*>(&Afrag[slot]) = pr;
        }
        __syncthreads();

        // 8 K-chunks of 32 -> 16 v_wmma_f32_16x16x32_bf16 per wave per j
#pragma unroll
        for (int kc = 0; kc < 8; ++kc) {
            const v16bf a =
                *reinterpret_cast<const v16bf*>(&Afrag[((wm * 8 + kc) * 32 + lane) * 16]);
            const __bf16* bp =
                W2t + ((size_t)(j * MDIM + kc * 32)) * MDIM + o0 + wn0 * 16 + (lane & 15);
            v16bf bf0, bf1;
#pragma unroll
            for (int s = 0; s < 16; ++s) {
                const int kb = (s < 8) ? (8 * (lane >> 4) + s) : (16 + 8 * (lane >> 4) + (s - 8));
                bf0[s] = bp[(size_t)kb * MDIM];
                bf1[s] = bp[(size_t)kb * MDIM + 16];
            }
            acc0 = __builtin_amdgcn_wmma_f32_16x16x32_bf16(false, a, false, bf0,
                                                           (short)0, acc0, false, false);
            acc1 = __builtin_amdgcn_wmma_f32_16x16x32_bf16(false, a, false, bf1,
                                                           (short)0, acc1, false, false);
        }
        // Stream next j-slab of B toward the caches (global_prefetch_b8).
        if (j + 1 < MDIM) {
            const __bf16* nxt = W2t + ((size_t)((j + 1) * MDIM) + t) * MDIM + o0;
            __builtin_prefetch(nxt, 0, 1);
            __builtin_prefetch(nxt + 64, 0, 1);
        }
        __syncthreads();
    }

    // Epilogue: C/D layout -> n = lane&15, m = (lane>>4)*8 + reg
    const int nl    = lane & 15;
    const int mbase = (lane >> 4) * 8;
#pragma unroll
    for (int ri = 0; ri < 8; ++ri) {
        const int brow = b0 + wm * 16 + mbase + ri;
        const int oc0  = o0 + wn0 * 16 + nl;
        const int oc1  = o0 + (wn0 + 1) * 16 + nl;
        cbuf[(size_t)brow * MDIM + oc0] = acc0[ri] + Kc[oc0];
        cbuf[(size_t)brow * MDIM + oc1] = acc1[ri] + Kc[oc1];
    }
}

// ---------------------------------------------------------------------------
// Kernel 6: out[b] = sum_o X[b,o] * (1 + c[b,o]) * Wf[o] + bf
// ---------------------------------------------------------------------------
__global__ void k_final(const float* __restrict__ X, const float* __restrict__ cbuf,
                        const float* __restrict__ Wf, const float* __restrict__ bfin,
                        float* __restrict__ out) {
    __shared__ float red[256];
    const int b = blockIdx.x, t = threadIdx.x;
    red[t] = X[b * MDIM + t] * (1.0f + cbuf[b * MDIM + t]) * Wf[t];
    __syncthreads();
    for (int off = 128; off; off >>= 1) { if (t < off) red[t] += red[t + off]; __syncthreads(); }
    if (t == 0) out[b] = red[0] + bfin[0];
}

// ---------------------------------------------------------------------------
extern "C" void kernel_launch(void* const* d_in, const int* in_sizes, int n_in,
                              void* d_out, int out_size, void* d_ws, size_t ws_size,
                              hipStream_t stream) {
    (void)in_sizes; (void)n_in; (void)out_size; (void)ws_size;
    const float* X     = (const float*)d_in[0];
    const float* W1    = (const float*)d_in[1];
    const float* b1    = (const float*)d_in[2];
    const float* gamma_= (const float*)d_in[3];
    const float* beta_ = (const float*)d_in[4];
    const float* W2    = (const float*)d_in[5];
    const float* b2    = (const float*)d_in[6];
    const float* Wf    = (const float*)d_in[7];
    const float* bfin  = (const float*)d_in[8];
    float* out = (float*)d_out;

    // Workspace layout: [W2t bf16 32MB][y2 512KB][g 1KB][h 1KB][Kc 1KB][cbuf 512KB]
    char* ws = (char*)d_ws;
    __bf16* W2t = (__bf16*)ws;
    float* fws  = (float*)(ws + (size_t)32 * 1024 * 1024);
    float* y2   = fws;
    float* g    = fws + 131072;
    float* h    = fws + 131072 + 256;
    float* Kc   = fws + 131072 + 512;
    float* cbuf = fws + 131072 + 768;

    k_y2     <<<BSZ, 256, 0, stream>>>(X, W1, b1, y2);
    k_moments<<<MDIM / 8, 256, 0, stream>>>(y2, gamma_, beta_, g, h);
    k_w2t    <<<KTOT / 32, 256, 0, stream>>>(W2, W2t);
    k_kc     <<<MDIM, 256, 0, stream>>>(W2, h, b2, Kc);
    k_gemm   <<<dim3(2, 16), 256, 0, stream>>>(y2, g, W2t, Kc, cbuf);
    k_final  <<<BSZ, 256, 0, stream>>>(X, cbuf, Wf, bfin, out);
}
